// TransformerAttention_31971736551773
// MI455X (gfx1250) — matrix operand order repaired and checked by reading, not compiled, not run on hardware
//
#include <hip/hip_runtime.h>
#include <hip/hip_bf16.h>

// ---------------------------------------------------------------------------
// Multi-head self-attention (B=4, C=512, N=2304, heads=8, d=64) for gfx1250.
// GEMMs: v_wmma_f32_16x16x32_bf16. Attention K/V staging: Tensor Data Mover
// (tensor_load_to_lds, double-buffered, TENSORcnt) with manual fallback.
// Workspace layout (bytes):
//   [0,        9437184)  Q   bf16 (b,h,n,d)
//   [9437184, 18874368)  K   bf16 (b,h,n,d)
//   [18874368,28311552)  V   bf16 (b,h,d,n)   <- d-major so TDM tiles are 2D
//   [28311552,37748736)  AO  bf16 (b,n,c)
// ---------------------------------------------------------------------------

typedef __bf16 bf16_t;
typedef __attribute__((ext_vector_type(16))) __bf16 v16bf;
typedef __attribute__((ext_vector_type(8)))  float  v8f;

#define BATCH  4
#define DIMC   512
#define SEQN   2304   // 48*48
#define NHEADS 8
#define HDIM   64
#define LDB    68     // bf16 tile leading dim (136B rows: 128B data + 8B pad)
#define NCHUNK (SEQN / 64)

#if defined(__gfx1250__) && __has_builtin(__builtin_amdgcn_tensor_load_to_lds)
#define USE_TDM 1
#else
#define USE_TDM 0
#endif

__device__ __forceinline__ v8f wmma_bf16(v16bf a, v16bf b, v8f c) {
    return __builtin_amdgcn_wmma_f32_16x16x32_bf16(
        /*neg_a=*/false, a, /*neg_b=*/false, b,
        /*c_mod=*/(short)0, c, /*reuse_a=*/false, /*reuse_b=*/false);
}

// A-matrix fragment (16x32 bf16, ISA 7.12.2): lane m = m0+lane%16, half = lane/16.
// VGPR r<4: K = 2r + 8*half ; r>=4: K = 16 + 2(r-4) + 8*half.
__device__ __forceinline__ v16bf load_a(const bf16_t* lds, int ld, int m0, int k0) {
    const int lane = threadIdx.x & 31;
    const int m    = m0 + (lane & 15);
    const int half = lane >> 4;
    v16bf a;
#pragma unroll
    for (int r = 0; r < 8; ++r) {
        const int kb = (r < 4) ? (2 * r + 8 * half) : (16 + 2 * (r - 4) + 8 * half);
        const bf16_t* p = lds + m * ld + k0 + kb;
        a[2 * r]     = p[0];
        a[2 * r + 1] = p[1];
    }
    return a;
}

// B-matrix fragment from LDS stored [n][k]: lane n = n0+lane%16; VGPR r<4: K = 2r + 8*half ; r>=4: K = 16 + 2(r-4) + 8*half.
__device__ __forceinline__ v16bf load_b(const bf16_t* lds, int ld, int n0, int k0) {
    const int lane = threadIdx.x & 31;
    const int n    = n0 + (lane & 15);
    const int half = lane >> 4;
    v16bf b;
#pragma unroll
    for (int r = 0; r < 8; ++r) {
        const bf16_t* p = lds + n * ld + k0 + ((r < 4) ? (2 * r + 8 * half) : (16 + 2 * (r - 4) + 8 * half));
        b[2 * r]     = p[0];
        b[2 * r + 1] = p[1];
    }
    return b;
}

// Butterfly reductions within each 16-lane half (wave32).
__device__ __forceinline__ float rowmax16(float v) {
    v = fmaxf(v, __shfl_xor(v, 1, 32));
    v = fmaxf(v, __shfl_xor(v, 2, 32));
    v = fmaxf(v, __shfl_xor(v, 4, 32));
    v = fmaxf(v, __shfl_xor(v, 8, 32));
    return v;
}
__device__ __forceinline__ float rowsum16(float v) {
    v += __shfl_xor(v, 1, 32);
    v += __shfl_xor(v, 2, 32);
    v += __shfl_xor(v, 4, 32);
    v += __shfl_xor(v, 8, 32);
    return v;
}

#if USE_TDM
typedef __attribute__((ext_vector_type(4))) unsigned int u32x4;
typedef __attribute__((ext_vector_type(8))) int          i32x8;
typedef __attribute__((ext_vector_type(4))) int          i32x4;

template <typename T>
__device__ __forceinline__ unsigned lds_offset(T* p) {
    return (unsigned)(unsigned long long)(__attribute__((address_space(3))) T*)p;
}

// DMA a 64x64 bf16 tile (contiguous dim0, element stride `stride` to next line)
// from global to LDS with +2-DWORD padding every 32 DWORDs (row stride 136B).
__device__ __forceinline__ void tdm_load_tile(unsigned lds_addr, const bf16_t* gp,
                                              unsigned stride) {
    const unsigned long long ga = (unsigned long long)gp;
    u32x4 g0;
    g0[0] = 1u;                                   // count=1, user descriptor
    g0[1] = lds_addr;                             // LDS byte address
    g0[2] = (unsigned)ga;                         // global_addr[31:0]
    g0[3] = (unsigned)(ga >> 32) | (2u << 30);    // global_addr[56:32] | type=2
    i32x8 g1;
    g1[0] = (1 << 16) | (1 << 20) | (4 << 22) | (1 << 25);
           // data_size=1(2B) | pad_enable | pad_interval=4(32dw) | pad_amount=1(2dw)
    g1[1] = (64 << 16);                           // tensor_dim0 = 64 (bits 79:48 lo)
    g1[2] = (64 << 16);                           // dim0 hi=0 | tensor_dim1 = 64 (lo)
    g1[3] = (64 << 16);                           // dim1 hi=0 | tile_dim0 = 64
    g1[4] = 64;                                   // tile_dim1 = 64, tile_dim2 = 0
    g1[5] = (int)stride;                          // tensor_dim0_stride[31:0]
    g1[6] = 0;                                    // stride hi | dim1_stride lo
    g1[7] = 0;
    const i32x4 z4 = {0, 0, 0, 0};                // groups 2/3 unused (2D tensor)
    const i32x8 z8 = {0, 0, 0, 0, 0, 0, 0, 0};
    __builtin_amdgcn_tensor_load_to_lds(g0, g1, z4, z4, z8, 0);  // 6-arg form
}
#endif

// ---------------------------------------------------------------------------
// Kernel 1: qkv = xf @ qkv_w^T + qkv_b ; scatter to Q/K (b,h,n,d), V (b,h,d,n).
// 64x64 tile, K-step 64 (4 WMMAs/wave between barriers). Grid (144,24), 256 thr.
// ---------------------------------------------------------------------------
__global__ __launch_bounds__(256) void qkv_gemm_kernel(
    const float* __restrict__ x, const float* __restrict__ w,
    const float* __restrict__ bias,
    bf16_t* __restrict__ qout, bf16_t* __restrict__ kout, bf16_t* __restrict__ vout)
{
    __shared__ bf16_t Al[64 * LDB];   // [row(n)][k]
    __shared__ bf16_t Bl[64 * LDB];   // [col][k]

    const int tid  = threadIdx.x;
    const int lane = tid & 31;
    const int wv   = tid >> 5;
    const int rowbase = blockIdx.x * 64;          // tile never crosses a batch
    const int colbase = blockIdx.y * 64;
    const int m0 = (wv >> 1) * 16;
    const int n0 = (wv & 1) * 32;
    const int bq    = rowbase / SEQN;
    const int nbase = rowbase % SEQN;
    const size_t xb = (size_t)bq * DIMC * SEQN + nbase;

    v8f c0 = {}; v8f c1 = {};

    for (int k0 = 0; k0 < DIMC; k0 += 64) {
#pragma unroll
        for (int it = 0; it < 4; ++it) {          // A: float4 over n (coalesced)
            const int idx = it * 256 + tid;
            const int kk = idx >> 4, r4 = (idx & 15) * 4;
            const float4 v4 = *(const float4*)(x + xb + (size_t)(k0 + kk) * SEQN + r4);
            if (it == 0 && k0 + 64 < DIMC)
                __builtin_prefetch(x + xb + (size_t)(k0 + 64 + kk) * SEQN + r4, 0, 1);
            Al[(r4 + 0) * LDB + kk] = (bf16_t)v4.x;
            Al[(r4 + 1) * LDB + kk] = (bf16_t)v4.y;
            Al[(r4 + 2) * LDB + kk] = (bf16_t)v4.z;
            Al[(r4 + 3) * LDB + kk] = (bf16_t)v4.w;
        }
#pragma unroll
        for (int it = 0; it < 4; ++it) {          // B: float4 over k (coalesced)
            const int idx = it * 256 + tid;
            const int cc = idx >> 4, k4 = (idx & 15) * 4;
            const float4 v4 = *(const float4*)(w + (size_t)(colbase + cc) * DIMC + k0 + k4);
            if (it == 0 && k0 + 64 < DIMC)
                __builtin_prefetch(w + (size_t)(colbase + cc) * DIMC + k0 + 64 + k4, 0, 1);
            Bl[cc * LDB + k4 + 0] = (bf16_t)v4.x;
            Bl[cc * LDB + k4 + 1] = (bf16_t)v4.y;
            Bl[cc * LDB + k4 + 2] = (bf16_t)v4.z;
            Bl[cc * LDB + k4 + 3] = (bf16_t)v4.w;
        }
        __syncthreads();

        const v16bf a0 = load_a(Al, LDB, m0, 0);
        const v16bf a1 = load_a(Al, LDB, m0, 32);
        c0 = wmma_bf16(a0, load_b(Bl, LDB, n0, 0),       c0);
        c0 = wmma_bf16(a1, load_b(Bl, LDB, n0, 32),      c0);
        c1 = wmma_bf16(a0, load_b(Bl, LDB, n0 + 16, 0),  c1);
        c1 = wmma_bf16(a1, load_b(Bl, LDB, n0 + 16, 32), c1);
        __syncthreads();
    }

    const int half = lane >> 4;
    const int cl   = lane & 15;
#pragma unroll
    for (int i = 0; i < 8; ++i) {
        const int n = nbase + m0 + i + 8 * half;
#pragma unroll
        for (int cc = 0; cc < 2; ++cc) {
            const int col = colbase + n0 + cc * 16 + cl;
            const float val = (cc ? c1[i] : c0[i]) + bias[col];
            const int which = col >> 9;           // 0=q,1=k,2=v
            const int cw = col & 511;
            const int h = cw >> 6, dd = cw & 63;
            if (which == 2)                       // V: (b,h,d,n)
                vout[((size_t)(bq * NHEADS + h) * HDIM + dd) * SEQN + n] = (bf16_t)val;
            else {
                bf16_t* dst = (which == 0) ? qout : kout;
                dst[(((size_t)bq * NHEADS + h) * SEQN + n) * HDIM + dd] = (bf16_t)val;
            }
        }
    }
}

// ---------------------------------------------------------------------------
// Kernel 2: flash attention. Grid (72 row-tiles, 8 heads, 4 batch), 256 thr.
// Wave w -> 16x16 tile: rows (w/4)*16, cols (w&3)*16 (keys for S, d for O).
// K/V chunks DMA'd by the TDM into double buffers, overlapped with compute.
// ---------------------------------------------------------------------------
__global__ __launch_bounds__(256) void attn_kernel(
    const bf16_t* __restrict__ Q, const bf16_t* __restrict__ K,
    const bf16_t* __restrict__ V, bf16_t* __restrict__ out)
{
    __shared__ bf16_t Ql[32 * LDB];      // [row][d]
    __shared__ bf16_t Kl[2][64 * LDB];   // [key][d]  (B-layout for S = Q K^T)
    __shared__ bf16_t Vl[2][64 * LDB];   // [d][key]  (B-layout for O = P V)
    __shared__ bf16_t Pl[32 * LDB];      // exp probs (A-layout rows)
    __shared__ float  maxpart[32][4], sumpart[32][4];
    __shared__ float  mrow[32], lrow[32], crow[32];

    const int tid  = threadIdx.x;
    const int lane = tid & 31;
    const int wv   = tid >> 5;
    const int m0 = (wv >> 2) * 16;
    const int n0 = (wv & 3) * 16;
    const int cg = wv & 3;
    const int b = blockIdx.z, h = blockIdx.y;
    const int rowbase = blockIdx.x * 32;
    const size_t bh  = ((size_t)b * NHEADS + h) * SEQN;   // Q/K row base (n-major)
    const size_t bhd = ((size_t)b * NHEADS + h) * HDIM;   // V row base   (d-major)
    const int half = lane >> 4;
    const int cl   = lane & 15;

#pragma unroll
    for (int it = 0; it < 2; ++it) {              // stage Q tile (32x64), 8B vectors
        const int idx = it * 256 + tid;
        const int r = idx >> 4, d4 = (idx & 15) * 4;
        const ushort4 qv = *(const ushort4*)(Q + (bh + rowbase + r) * HDIM + d4);
        *(ushort4*)(reinterpret_cast<unsigned short*>(Ql) + r * LDB + d4) = qv;
    }
    if (tid < 32) { mrow[tid] = -1e30f; lrow[tid] = 0.0f; }
#if USE_TDM
    if (tid < 32) {                               // wave 0: DMA chunk 0 (async)
        tdm_load_tile(lds_offset(&Kl[0][0]), K + bh * HDIM, HDIM);
        tdm_load_tile(lds_offset(&Vl[0][0]), V + bhd * SEQN, SEQN);
    }
#endif
    __syncthreads();

    const v16bf aq0 = load_a(Ql, LDB, m0, 0);     // Q frags live in regs all loop
    const v16bf aq1 = load_a(Ql, LDB, m0, 32);
    v8f oacc = {};

    for (int c = 0; c < NCHUNK; ++c) {
        const int j0  = c * 64;
        const int buf = c & 1;
#if USE_TDM
        if (tid < 32) {                           // wave 0: issue c+1, wait for c
            if (c + 1 < NCHUNK) {
                tdm_load_tile(lds_offset(&Kl[buf ^ 1][0]),
                              K + (bh + j0 + 64) * HDIM, HDIM);
                tdm_load_tile(lds_offset(&Vl[buf ^ 1][0]),
                              V + bhd * SEQN + (j0 + 64), SEQN);
                __builtin_amdgcn_s_wait_tensorcnt((short)2);
            } else {
                __builtin_amdgcn_s_wait_tensorcnt((short)0);
            }
        }
        __syncthreads();
#else
#pragma unroll
        for (int it = 0; it < 4; ++it) {          // manual staging (both row-major)
            const int idx = it * 256 + tid;
            const int r = idx >> 4, k4 = (idx & 15) * 4;
            const ushort4 kv = *(const ushort4*)(K + (bh + j0 + r) * HDIM + k4);
            const ushort4 vv = *(const ushort4*)(V + (bhd + r) * SEQN + j0 + k4);
            if (it == 0 && j0 + 64 < SEQN) {
                __builtin_prefetch(K + (bh + j0 + 64 + r) * HDIM + k4, 0, 1);
                __builtin_prefetch(V + (bhd + r) * SEQN + j0 + 64 + k4, 0, 1);
            }
            *(ushort4*)(reinterpret_cast<unsigned short*>(Kl[buf]) + r * LDB + k4) = kv;
            *(ushort4*)(reinterpret_cast<unsigned short*>(Vl[buf]) + r * LDB + k4) = vv;
        }
        __syncthreads();
#endif

        v8f s = {};                               // S tile (16x16 per wave)
        s = wmma_bf16(aq0, load_b(Kl[buf], LDB, n0, 0),  s);
        s = wmma_bf16(aq1, load_b(Kl[buf], LDB, n0, 32), s);
#pragma unroll
        for (int i = 0; i < 8; ++i) s[i] *= 0.125f;   // 1/sqrt(64)

        // wave-local row max over this wave's 16 key-cols
#pragma unroll
        for (int i = 0; i < 8; ++i) {
            const float mx = rowmax16(s[i]);
            if (cl == 0) maxpart[m0 + i + 8 * half][cg] = mx;
        }
        __syncthreads();

        if (tid < 32) {                           // merge 4 parts, update running max
            const float cmax = fmaxf(fmaxf(maxpart[tid][0], maxpart[tid][1]),
                                     fmaxf(maxpart[tid][2], maxpart[tid][3]));
            const float nm = fmaxf(mrow[tid], cmax);
            crow[tid] = __expf(mrow[tid] - nm);
            mrow[tid] = nm;
        }
        __syncthreads();

        // p = exp(s - m), fully parallel; write P (bf16) + partial row sums
#pragma unroll
        for (int i = 0; i < 8; ++i) {
            const int rl = m0 + i + 8 * half;
            const float p = __expf(s[i] - mrow[rl]);
            Pl[rl * LDB + n0 + cl] = (bf16_t)p;
            const float ps = rowsum16(p);
            if (cl == 0) sumpart[rl][cg] = ps;
        }
        __syncthreads();

        if (tid < 32)                             // running denominator
            lrow[tid] = lrow[tid] * crow[tid] +
                        (sumpart[tid][0] + sumpart[tid][1] +
                         sumpart[tid][2] + sumpart[tid][3]);

        // rescale O, accumulate P @ V
#pragma unroll
        for (int i = 0; i < 8; ++i) oacc[i] *= crow[m0 + i + 8 * half];
        oacc = wmma_bf16(load_a(Pl, LDB, m0, 0),  load_b(Vl[buf], LDB, n0, 0),  oacc);
        oacc = wmma_bf16(load_a(Pl, LDB, m0, 32), load_b(Vl[buf], LDB, n0, 32), oacc);
        __syncthreads();                          // release buf / Pl for next chunk
    }

#pragma unroll
    for (int i = 0; i < 8; ++i) {
        const int rl = m0 + i + 8 * half;
        const int n  = rowbase + rl;
        const int c  = h * HDIM + n0 + cl;
        out[((size_t)b * SEQN + n) * DIMC + c] = (bf16_t)(oacc[i] / lrow[rl]);
    }
}

// ---------------------------------------------------------------------------
// Kernel 3: out = AO @ proj_w^T + proj_b, transposed to (B,C,N) fp32 via an
// LDS C-tile so global writes are coalesced. Grid (144, 8), 256 thr.
// ---------------------------------------------------------------------------
__global__ __launch_bounds__(256) void proj_gemm_kernel(
    const bf16_t* __restrict__ A, const float* __restrict__ w,
    const float* __restrict__ bias, float* __restrict__ out)
{
    __shared__ bf16_t Al[64 * LDB];
    __shared__ bf16_t Bl[64 * LDB];
    __shared__ float  Cl[64 * 65];    // [col][n]

    const int tid  = threadIdx.x;
    const int lane = tid & 31;
    const int wv   = tid >> 5;
    const int rowbase = blockIdx.x * 64;
    const int colbase = blockIdx.y * 64;
    const int m0 = (wv >> 1) * 16;
    const int n0 = (wv & 1) * 32;
    const int bq    = rowbase / SEQN;
    const int nbase = rowbase % SEQN;

    v8f c0 = {}; v8f c1 = {};

    for (int k0 = 0; k0 < DIMC; k0 += 64) {
#pragma unroll
        for (int it = 0; it < 4; ++it) {          // A (bf16): ushort4 over k
            const int idx = it * 256 + tid;
            const int r = idx >> 4, k4 = (idx & 15) * 4;
            const ushort4 a4 = *(const ushort4*)(A + (size_t)(rowbase + r) * DIMC + k0 + k4);
            if (it == 0 && k0 + 64 < DIMC)
                __builtin_prefetch(A + (size_t)(rowbase + r) * DIMC + k0 + 64 + k4, 0, 1);
            *(ushort4*)(reinterpret_cast<unsigned short*>(Al) + r * LDB + k4) = a4;
        }
#pragma unroll
        for (int it = 0; it < 4; ++it) {          // B (fp32): float4 over k
            const int idx = it * 256 + tid;
            const int cc = idx >> 4, k4 = (idx & 15) * 4;
            const float4 v4 = *(const float4*)(w + (size_t)(colbase + cc) * DIMC + k0 + k4);
            if (it == 0 && k0 + 64 < DIMC)
                __builtin_prefetch(w + (size_t)(colbase + cc) * DIMC + k0 + 64 + k4, 0, 1);
            Bl[cc * LDB + k4 + 0] = (bf16_t)v4.x;
            Bl[cc * LDB + k4 + 1] = (bf16_t)v4.y;
            Bl[cc * LDB + k4 + 2] = (bf16_t)v4.z;
            Bl[cc * LDB + k4 + 3] = (bf16_t)v4.w;
        }
        __syncthreads();

        const v16bf a0 = load_a(Al, LDB, m0, 0);
        const v16bf a1 = load_a(Al, LDB, m0, 32);
        c0 = wmma_bf16(a0, load_b(Bl, LDB, n0, 0),       c0);
        c0 = wmma_bf16(a1, load_b(Bl, LDB, n0, 32),      c0);
        c1 = wmma_bf16(a0, load_b(Bl, LDB, n0 + 16, 0),  c1);
        c1 = wmma_bf16(a1, load_b(Bl, LDB, n0 + 16, 32), c1);
        __syncthreads();
    }

    const int half = lane >> 4;
    const int cl   = lane & 15;
#pragma unroll
    for (int i = 0; i < 8; ++i) {                 // C tile -> LDS (add bias)
        const int rl = m0 + i + 8 * half;
#pragma unroll
        for (int cc = 0; cc < 2; ++cc) {
            const int col = n0 + cc * 16 + cl;
            Cl[col * 65 + rl] = (cc ? c1[i] : c0[i]) + bias[colbase + col];
        }
    }
    __syncthreads();
#pragma unroll
    for (int it = 0; it < 16; ++it) {             // coalesced transposed store
        const int idx = it * 256 + tid;
        const int col = idx >> 6, nn = idx & 63;
        out[(size_t)bq * DIMC * SEQN + (size_t)(colbase + col) * SEQN + nbase + nn] =
            Cl[col * 65 + nn];
    }
}

// ---------------------------------------------------------------------------
extern "C" void kernel_launch(void* const* d_in, const int* in_sizes, int n_in,
                              void* d_out, int out_size, void* d_ws, size_t ws_size,
                              hipStream_t stream) {
    const float* x      = (const float*)d_in[0];
    const float* qkv_w  = (const float*)d_in[1];
    const float* qkv_b  = (const float*)d_in[2];
    const float* proj_w = (const float*)d_in[3];
    const float* proj_b = (const float*)d_in[4];
    float*       out    = (float*)d_out;

    char* ws = (char*)d_ws;
    bf16_t* qb = (bf16_t*)(ws);
    bf16_t* kb = (bf16_t*)(ws + 9437184);
    bf16_t* vb = (bf16_t*)(ws + 18874368);
    bf16_t* ab = (bf16_t*)(ws + 28311552);

    qkv_gemm_kernel<<<dim3(144, 24), 256, 0, stream>>>(x, qkv_w, qkv_b, qb, kb, vb);
    attn_kernel<<<dim3(72, NHEADS, BATCH), 256, 0, stream>>>(qb, kb, vb, ab);
    proj_gemm_kernel<<<dim3(144, 8), 256, 0, stream>>>(ab, proj_w, proj_b, out);
}
